// LSTMModel_64682207477837
// MI455X (gfx1250) — matrix operand order repaired and checked by reading, not compiled, not run on hardware
//
#include <hip/hip_runtime.h>
#include <hip/hip_bf16.h>
#include <math.h>

// Problem constants (from reference)
#define S_   4096
#define L_   12
#define CE_  64
#define CH_  128
#define WE_  256
#define H_   512
#define TAGS_ 64
#define G1_  (4*CH_)   // 512  char gate width
#define G2_  (4*H_)    // 2048 word gate width
#define K1_  (CE_+CH_) // 192  char concat K
#define K2_  (CH_+WE_) // 384  word concat K
#define NB_  16        // persistent blocks for word recurrence

typedef __attribute__((ext_vector_type(16))) _Float16 v16h;
typedef __attribute__((ext_vector_type(8)))  float    v8f;

__device__ __forceinline__ float sigf(float x) { return 1.0f / (1.0f + expf(-x)); }

// ---- WMMA fragment index helpers (wave32, 16x16x32 f16, per ISA 7.12.2) ----
// A (16x32, MxK): lane&15 = M ; element e: r=e>>1,
//   k = ((r>>2)<<4) + ((r&3)<<1) + (laneHi?8:0) + (e&1)
// B (32x16, KxN): lane&15 = N ; element e: k = a_k(e, laneHi), same order as A
// C/D (16x16 f32): VGPR r: M = r + (laneHi?8:0), N = lane&15
__device__ __forceinline__ int a_k(int e, int hi) {
    int r = e >> 1;
    return ((r >> 2) << 4) + ((r & 3) << 1) + (hi ? 8 : 0) + (e & 1);
}

// ========================= init kernels =========================
__global__ void init_small_kernel(const float* bih1, const float* bhh1,
                                  const float* bih2, const float* bhh2,
                                  const float* hx0,
                                  float* b1, float* b2, float* hbuf, unsigned* gen) {
    int t = threadIdx.x;
    for (int i = t; i < G1_; i += 256) b1[i] = bih1[i] + bhh1[i];
    for (int i = t; i < G2_; i += 256) b2[i] = bih2[i] + bhh2[i];
    for (int i = t; i < H_;  i += 256) { hbuf[i] = hx0[i]; hbuf[H_ + i] = 0.0f; }
    if (t == 0) *gen = 0u;
}

__global__ void init_state_kernel(const float* c_hx0, const float* c_cx0,
                                  float* H1, float* C1) {
    int idx = blockIdx.x * blockDim.x + threadIdx.x;  // S_*CH_ threads
    int j = idx & (CH_ - 1);
    H1[idx] = c_hx0[j];
    C1[idx] = c_cx0[j];
}

// ============== char LSTM: gate GEMM  G = [x_t | h] @ [Wih1;Whh1]^T ==============
// M=4096, N=512, K=192.  One 16x16 tile per wave.
__global__ void char_gemm_kernel(int t, const int* c_seq, const float* char_emb,
                                 const float* H1, const float* Wih1, const float* Whh1,
                                 float* G) {
    int wave = (blockIdx.x * blockDim.x + threadIdx.x) >> 5;
    int lane = threadIdx.x & 31;
    int hi = lane >> 4;
    const int NT = G1_ / 16;              // 32 n-tiles
    int mt = wave / NT, nt = wave % NT;   // mt 0..255
    int m = mt * 16 + (lane & 15);
    int n = nt * 16 + (lane & 15);
    int cidx = c_seq[m * L_ + t];

    v8f acc = {};
    for (int kc = 0; kc < K1_ / 32; ++kc) {
        v16h a, b;
#pragma unroll
        for (int e = 0; e < 16; ++e) {
            int ka = a_k(e, hi) + kc * 32;
            float av = (ka < CE_) ? char_emb[cidx * CE_ + ka]
                                  : H1[m * CH_ + (ka - CE_)];
            a[e] = (_Float16)av;
            int kb = a_k(e, hi) + kc * 32;
            float bv = (kb < CE_) ? Wih1[n * CE_ + kb]
                                  : Whh1[n * CH_ + (kb - CE_)];
            b[e] = (_Float16)bv;
        }
        acc = __builtin_amdgcn_wmma_f32_16x16x32_f16(false, a, false, b,
                                                     (short)0, acc, false, false);
    }
#pragma unroll
    for (int r = 0; r < 8; ++r) {
        int mm = mt * 16 + r + (hi ? 8 : 0);
        G[mm * G1_ + n] = acc[r];
    }
}

// Elementwise char-LSTM gate update (PyTorch gate order i,f,g,o)
__global__ void char_update_kernel(const float* G, const float* b1,
                                   float* H1, float* C1) {
    int idx = blockIdx.x * blockDim.x + threadIdx.x;  // S_*CH_
    int s = idx >> 7, j = idx & (CH_ - 1);
    float gi = G[s * G1_ + j]            + b1[j];
    float gf = G[s * G1_ + CH_ + j]      + b1[CH_ + j];
    float gg = G[s * G1_ + 2 * CH_ + j]  + b1[2 * CH_ + j];
    float go = G[s * G1_ + 3 * CH_ + j]  + b1[3 * CH_ + j];
    float c = sigf(gf) * C1[idx] + sigf(gi) * tanhf(gg);
    float h = sigf(go) * tanhf(c);
    C1[idx] = c;
    H1[idx] = h;
}

// ========= word LSTM input precompute: XG2 = [char_rep | wv] @ Wih2^T =========
// M=4096, N=2048, K=384
__global__ void xg2_gemm_kernel(const int* w_seq, const float* word_emb,
                                const float* H1, const float* Wih2, float* XG2) {
    int wave = (blockIdx.x * blockDim.x + threadIdx.x) >> 5;
    int lane = threadIdx.x & 31;
    int hi = lane >> 4;
    const int NT = G2_ / 16;              // 128 n-tiles
    int mt = wave / NT, nt = wave % NT;
    int m = mt * 16 + (lane & 15);
    int n = nt * 16 + (lane & 15);
    int widx = w_seq[m];

    v8f acc = {};
    for (int kc = 0; kc < K2_ / 32; ++kc) {
        v16h a, b;
#pragma unroll
        for (int e = 0; e < 16; ++e) {
            int ka = a_k(e, hi) + kc * 32;
            float av = (ka < CH_) ? H1[m * CH_ + ka]
                                  : word_emb[widx * WE_ + (ka - CH_)];
            a[e] = (_Float16)av;
            int kb = a_k(e, hi) + kc * 32;
            b[e] = (_Float16)Wih2[n * K2_ + kb];
        }
        acc = __builtin_amdgcn_wmma_f32_16x16x32_f16(false, a, false, b,
                                                     (short)0, acc, false, false);
    }
#pragma unroll
    for (int r = 0; r < 8; ++r) {
        int mm = mt * 16 + r + (hi ? 8 : 0);
        XG2[mm * G2_ + n] = acc[r];
    }
}

// ============== word LSTM recurrence: persistent NB_ blocks ==============
// Block b owns h indices [b*32, b*32+32) -> 128 Whh2 rows (4 gates x 32).
// One grid barrier per step (monotonic gen counter), double-buffered h.
__global__ void __launch_bounds__(256)
word_rec_kernel(const float* XG2, const float* Whh2, const float* b2,
                const float* hx0, const float* cx0,
                float* hbuf, float* outs, unsigned* gen) {
    __shared__ float sh_h[H_];
    __shared__ float sh_ps[256];
    __shared__ float sh_gate[128];
    __shared__ float sh_c[32];

    int tid = threadIdx.x;
    int base = blockIdx.x * 32;
    int d = tid >> 1;                 // 0..127: which gate row
    int p = tid & 1;                  // which half of the K=512 dot
    int q = d >> 5;                   // gate index 0..3 (i,f,g,o)
    int row = q * H_ + base + (d & 31);

    for (int i = tid; i < H_; i += 256) sh_h[i] = hx0[i];
    if (tid < 32) sh_c[tid] = cx0[base + tid];
    __syncthreads();

    const float4* wr4 = (const float4*)(Whh2 + row * H_ + p * 256);

    for (int s = 0; s < S_; ++s) {
        // partial dot: 256 elements per thread
        const float* hh = sh_h + p * 256;
        float sum = 0.0f;
#pragma unroll 8
        for (int k = 0; k < 64; ++k) {
            float4 w = wr4[k];
            sum += w.x * hh[4 * k] + w.y * hh[4 * k + 1] +
                   w.z * hh[4 * k + 2] + w.w * hh[4 * k + 3];
        }
        sh_ps[tid] = sum;
        __syncthreads();

        if (tid < 128) {
            int qq = tid >> 5;
            int rr = qq * H_ + base + (tid & 31);
            sh_gate[tid] = sh_ps[2 * tid] + sh_ps[2 * tid + 1] +
                           XG2[s * G2_ + rr] + b2[rr];
        }
        __syncthreads();

        if (tid < 32) {
            float gi = sh_gate[tid];
            float gf = sh_gate[32 + tid];
            float gg = sh_gate[64 + tid];
            float go = sh_gate[96 + tid];
            float c = sigf(gf) * sh_c[tid] + sigf(gi) * tanhf(gg);
            float h = sigf(go) * tanhf(c);
            sh_c[tid] = c;
            int hidx = base + tid;
            hbuf[((s + 1) & 1) * H_ + hidx] = h;
            outs[s * H_ + hidx] = h;
        }
        __syncthreads();

        // grid barrier: each block bumps gen once; wait for all NB_ arrivals
        if (tid == 0) {
            __threadfence();
            __hip_atomic_fetch_add(gen, 1u, __ATOMIC_ACQ_REL, __HIP_MEMORY_SCOPE_AGENT);
            unsigned target = (unsigned)(s + 1) * NB_;
            while (__hip_atomic_load(gen, __ATOMIC_ACQUIRE, __HIP_MEMORY_SCOPE_AGENT) < target)
                __builtin_amdgcn_s_sleep(1);
        }
        __syncthreads();

        // stage the full new h for next step
        const float* hb = hbuf + ((s + 1) & 1) * H_;
        for (int i = tid; i < H_; i += 256) sh_h[i] = hb[i];
        __syncthreads();
    }
}

// ================ tag projection: tag = outs @ W_out^T + b_out ================
// M=4096, N=64, K=512
__global__ void tag_gemm_kernel(const float* outs, const float* W_out,
                                const float* b_out, float* tag) {
    int wave = (blockIdx.x * blockDim.x + threadIdx.x) >> 5;
    int lane = threadIdx.x & 31;
    int hi = lane >> 4;
    const int NT = TAGS_ / 16;            // 4 n-tiles
    int mt = wave / NT, nt = wave % NT;
    int m = mt * 16 + (lane & 15);
    int n = nt * 16 + (lane & 15);

    v8f acc = {};
    for (int kc = 0; kc < H_ / 32; ++kc) {
        v16h a, b;
#pragma unroll
        for (int e = 0; e < 16; ++e) {
            int ka = a_k(e, hi) + kc * 32;
            a[e] = (_Float16)outs[m * H_ + ka];
            int kb = a_k(e, hi) + kc * 32;
            b[e] = (_Float16)W_out[n * H_ + kb];
        }
        acc = __builtin_amdgcn_wmma_f32_16x16x32_f16(false, a, false, b,
                                                     (short)0, acc, false, false);
    }
    float bo = b_out[n];
#pragma unroll
    for (int r = 0; r < 8; ++r) {
        int mm = mt * 16 + r + (hi ? 8 : 0);
        tag[mm * TAGS_ + n] = acc[r] + bo;
    }
}

// ======================= row-wise log_softmax (64 cols) =======================
__global__ void log_softmax_kernel(const float* tag, float* out) {
    int wave = (blockIdx.x * blockDim.x + threadIdx.x) >> 5;  // row
    int lane = threadIdx.x & 31;
    float x0 = tag[wave * TAGS_ + lane];
    float x1 = tag[wave * TAGS_ + 32 + lane];
    float m = fmaxf(x0, x1);
#pragma unroll
    for (int o = 16; o > 0; o >>= 1) m = fmaxf(m, __shfl_xor(m, o, 32));
    float sum = expf(x0 - m) + expf(x1 - m);
#pragma unroll
    for (int o = 16; o > 0; o >>= 1) sum += __shfl_xor(sum, o, 32);
    float l = m + logf(sum);
    out[wave * TAGS_ + lane] = x0 - l;
    out[wave * TAGS_ + 32 + lane] = x1 - l;
}

// ================================ launcher ================================
extern "C" void kernel_launch(void* const* d_in, const int* in_sizes, int n_in,
                              void* d_out, int out_size, void* d_ws, size_t ws_size,
                              hipStream_t stream) {
    const int*   w_seq    = (const int*)d_in[0];
    const int*   c_seq    = (const int*)d_in[1];
    const float* char_emb = (const float*)d_in[2];
    const float* word_emb = (const float*)d_in[3];
    const float* c_hx0    = (const float*)d_in[4];
    const float* c_cx0    = (const float*)d_in[5];
    const float* hx0      = (const float*)d_in[6];
    const float* cx0      = (const float*)d_in[7];
    const float* Wih1     = (const float*)d_in[8];
    const float* Whh1     = (const float*)d_in[9];
    const float* bih1     = (const float*)d_in[10];
    const float* bhh1     = (const float*)d_in[11];
    const float* Wih2     = (const float*)d_in[12];
    const float* Whh2     = (const float*)d_in[13];
    const float* bih2     = (const float*)d_in[14];
    const float* bhh2     = (const float*)d_in[15];
    const float* W_out    = (const float*)d_in[16];
    const float* b_out    = (const float*)d_in[17];
    float* out = (float*)d_out;

    // workspace layout (floats)
    float* W = (float*)d_ws;
    size_t o = 0;
    float* G    = W + o; o += (size_t)S_ * G1_;   // 4096x512 gate scratch
    float* H1   = W + o; o += (size_t)S_ * CH_;   // char h state / char_rep
    float* C1   = W + o; o += (size_t)S_ * CH_;   // char c state
    float* XG2  = W + o; o += (size_t)S_ * G2_;   // word input-gate precompute
    float* OUTS = W + o; o += (size_t)S_ * H_;    // word LSTM outputs
    float* TAG  = W + o; o += (size_t)S_ * TAGS_; // tag_space
    float* B1   = W + o; o += G1_;
    float* B2   = W + o; o += G2_;
    float* HB   = W + o; o += 2 * H_;             // double-buffered h
    unsigned* GEN = (unsigned*)(W + o);

    // init
    init_small_kernel<<<1, 256, 0, stream>>>(bih1, bhh1, bih2, bhh2, hx0, B1, B2, HB, GEN);
    init_state_kernel<<<(S_ * CH_) / 256, 256, 0, stream>>>(c_hx0, c_cx0, H1, C1);

    // char LSTM: 12 steps of (WMMA gate GEMM + elementwise update)
    const int charWaves = (S_ / 16) * (G1_ / 16);            // 8192
    for (int t = 0; t < L_; ++t) {
        char_gemm_kernel<<<charWaves * 32 / 128, 128, 0, stream>>>(
            t, c_seq, char_emb, H1, Wih1, Whh1, G);
        char_update_kernel<<<(S_ * CH_) / 256, 256, 0, stream>>>(G, B1, H1, C1);
    }

    // word LSTM input gates (fully parallel WMMA GEMM)
    const int xg2Waves = (S_ / 16) * (G2_ / 16);             // 32768
    xg2_gemm_kernel<<<xg2Waves * 32 / 128, 128, 0, stream>>>(w_seq, word_emb, H1, Wih2, XG2);

    // sequential word recurrence: persistent 16-block kernel
    word_rec_kernel<<<NB_, 256, 0, stream>>>(XG2, Whh2, B2, hx0, cx0, HB, OUTS, GEN);

    // output projection + log_softmax
    const int tagWaves = (S_ / 16) * (TAGS_ / 16);           // 1024
    tag_gemm_kernel<<<tagWaves * 32 / 128, 128, 0, stream>>>(OUTS, W_out, b_out, TAG);
    log_softmax_kernel<<<(S_ * 32) / 128, 128, 0, stream>>>(TAG, out);
}
